// Regressor_27006754357809
// MI455X (gfx1250) — matrix operand order repaired and checked by reading, not compiled, not run on hardware
//
#include <hip/hip_runtime.h>
#include <hip/hip_fp16.h>

typedef __attribute__((ext_vector_type(16))) _Float16 v16h;
typedef __attribute__((ext_vector_type(8)))  float    v8f;

#define HIDDEN  16
#define IN_SZ   64
#define SEQ     12
#define BATCH   131072
#define WPB     8                 // waves per block
#define THREADS (WPB * 32)
#define LDS_STRIDE 24             // halves per h-row: 16 data + 8 pad (48B, 16B-aligned, conflict-free)

__device__ __forceinline__ float fast_tanh(float x) {
#if __has_builtin(__builtin_amdgcn_tanhf)
  return __builtin_amdgcn_tanhf(x);          // gfx1250 V_TANH_F32
#elif __has_builtin(__builtin_amdgcn_tanh_f32)
  return __builtin_amdgcn_tanh_f32(x);
#else
  return tanhf(x);
#endif
}
__device__ __forceinline__ float fast_sigmoid(float x) {
  return 0.5f * fast_tanh(0.5f * x) + 0.5f;  // sigmoid(x) = (1+tanh(x/2))/2
}

// Build a v16h from two contiguous 8-float runs (elements 0..7 <- p, 8..15 <- q).
// Matches the 16-bit WMMA A/B VGPR layouts when p/q are chosen per-lane.
__device__ __forceinline__ v16h cvt8_8(const float* __restrict__ p,
                                       const float* __restrict__ q) {
  float4 a = *(const float4*)(p);
  float4 b = *(const float4*)(p + 4);
  float4 c = *(const float4*)(q);
  float4 d = *(const float4*)(q + 4);
  v16h v;
  v[0]=(_Float16)a.x;  v[1]=(_Float16)a.y;  v[2]=(_Float16)a.z;  v[3]=(_Float16)a.w;
  v[4]=(_Float16)b.x;  v[5]=(_Float16)b.y;  v[6]=(_Float16)b.z;  v[7]=(_Float16)b.w;
  v[8]=(_Float16)c.x;  v[9]=(_Float16)c.y;  v[10]=(_Float16)c.z; v[11]=(_Float16)c.w;
  v[12]=(_Float16)d.x; v[13]=(_Float16)d.y; v[14]=(_Float16)d.z; v[15]=(_Float16)d.w;
  return v;
}
// 16 contiguous floats -> elements 0..15 (B-operand per-lane run)
__device__ __forceinline__ v16h cvtB16(const float* __restrict__ p) {
  return cvt8_8(p, p + 8);
}

__global__ void __launch_bounds__(THREADS)
lstm_wmma_kernel(const float* __restrict__ x,
                 const float* __restrict__ W_ih,
                 const float* __restrict__ W_hh,
                 const float* __restrict__ b_ih,
                 const float* __restrict__ b_hh,
                 const float* __restrict__ fc_w,
                 const float* __restrict__ fc_b,
                 float* __restrict__ out) {
  // per-wave h staging tile: [16 rows][LDS_STRIDE halves]
  __shared__ __align__(16) _Float16 lds_h[WPB][16 * LDS_STRIDE];

  const int tid   = threadIdx.x;
  const int wave  = tid >> 5;
  const int lane  = tid & 31;
  const int laneN = lane & 15;       // M (A-rows) or N (B-cols / C-cols)
  const int hi    = lane >> 4;       // which K-half this lane carries

  const long tile = (long)blockIdx.x * WPB + wave;   // 16 batch rows per wave
  const int  row  = (int)tile * 16 + laneN;          // batch row this lane loads (A: M = lane%16)

  // ---- Weights as WMMA B-operands (K x N = 32 x 16, f16) ----
  // B k-order used here: lane L holds column N = L%16, elements e=0..15 are K = e + 16*(L>=16).
  v16h Bih[2][4];                     // [K-tile k][gate tile n]; gate g = 16n + N
  #pragma unroll
  for (int k = 0; k < 2; ++k)
    #pragma unroll
    for (int n = 0; n < 4; ++n)
      Bih[k][n] = cvtB16(W_ih + (size_t)(n * 16 + laneN) * IN_SZ + k * 32 + hi * 16);

  v16h Bhh[4];
  #pragma unroll
  for (int n = 0; n < 4; ++n) { v16h z = {}; Bhh[n] = z; }
  if (hi == 0) {                      // K = 0..15 real hidden, K = 16..31 zero pad
    #pragma unroll
    for (int n = 0; n < 4; ++n)
      Bhh[n] = cvtB16(W_hh + (size_t)(n * 16 + laneN) * HIDDEN);
  }

  // Bias splat: C/D element (VGPR r, lane L) has N = L%16 -> gate g = 16n + N
  float bias[4];
  #pragma unroll
  for (int n = 0; n < 4; ++n)
    bias[n] = b_ih[n * 16 + laneN] + b_hh[n * 16 + laneN];

  // ---- State ----
  v8f c = {};                                      // cell state, C/D layout
  union { v16h v; uint4 q[2]; } ah;                // h as A-operand (K 16..31 stay zero)
  ah.v = v16h{};

  // A-operand x base: lane carries row M=laneN, K-half hi.
  // A k-order used here (same as B): elements 0..15 -> K = 16*hi + e.
  const float* xbase = x + (size_t)row * (SEQ * IN_SZ) + hi * 16;
  _Float16* hrow = &lds_h[wave][0];

  for (int t = 0; t < SEQ; ++t) {
    const float* xr = xbase + t * IN_SZ;
    if (t + 1 < SEQ) __builtin_prefetch(xr + IN_SZ, 0, 3);   // global_prefetch_b8

    v16h a0 = cvt8_8(xr,      xr + 8);    // K-tile 0: K = 0..31
    v16h a1 = cvt8_8(xr + 32, xr + 40);   // K-tile 1: K = 32..63

    // gates[n] = x_t @ W_ih^T (tile n) + h @ W_hh^T (tile n) + bias  -- all 16x16, C/D f32
    v8f g4[4];
    #pragma unroll
    for (int n = 0; n < 4; ++n) {
      v8f acc = { bias[n], bias[n], bias[n], bias[n], bias[n], bias[n], bias[n], bias[n] };
      acc = __builtin_amdgcn_wmma_f32_16x16x32_f16(false, a0,   false, Bih[0][n], (short)0, acc, false, false);
      acc = __builtin_amdgcn_wmma_f32_16x16x32_f16(false, a1,   false, Bih[1][n], (short)0, acc, false, false);
      acc = __builtin_amdgcn_wmma_f32_16x16x32_f16(false, ah.v, false, Bhh[n],    (short)0, acc, false, false);
      g4[n] = acc;
    }

    // LSTM pointwise: gate tile n==0:i 1:f 2:g 3:o, all congruent with c's C/D layout.
    // C/D layout: VGPR r, lane L -> M = r + 8*(L/16) (batch-in-tile), N = L%16 (hidden).
    #pragma unroll
    for (int r = 0; r < 8; ++r) {
      float iv = fast_sigmoid(g4[0][r]);
      float fv = fast_sigmoid(g4[1][r]);
      float gv = fast_tanh(g4[2][r]);
      float ov = fast_sigmoid(g4[3][r]);
      float cv = fv * c[r] + iv * gv;
      c[r] = cv;
      float hv = ov * fast_tanh(cv);
      hrow[(r + 8 * hi) * LDS_STRIDE + laneN] = (_Float16)hv;   // h[M][K=N], f16
    }

    // C-layout -> A-layout transpose through LDS (in-order per wave; belt & suspenders wait)
    asm volatile("s_wait_dscnt 0" ::: "memory");
    if (hi == 0) { ah.q[0] = *(const uint4*)(&hrow[laneN * LDS_STRIDE]); ah.q[1] = *(const uint4*)(&hrow[laneN * LDS_STRIDE + 8]); }  // hi=0: 16 halves K = 0..15; hi=1 stays zero
  }

  // Head: out[b] = h_last[b] . fc_w + fc_b ; lanes 0..15 each own one row M.
  asm volatile("s_wait_dscnt 0" ::: "memory");
  if (hi == 0) {
    const _Float16* hr = &lds_h[wave][laneN * LDS_STRIDE];
    float acc = fc_b[0];
    #pragma unroll
    for (int k = 0; k < HIDDEN; ++k) acc += (float)hr[k] * fc_w[k];
    out[tile * 16 + laneN] = acc;
  }
}

extern "C" void kernel_launch(void* const* d_in, const int* in_sizes, int n_in,
                              void* d_out, int out_size, void* d_ws, size_t ws_size,
                              hipStream_t stream) {
  const float* x    = (const float*)d_in[0];
  const float* W_ih = (const float*)d_in[1];
  const float* W_hh = (const float*)d_in[2];
  const float* b_ih = (const float*)d_in[3];
  const float* b_hh = (const float*)d_in[4];
  const float* fc_w = (const float*)d_in[5];
  const float* fc_b = (const float*)d_in[6];
  float* out = (float*)d_out;

  dim3 grid(BATCH / (16 * WPB));   // 1024 blocks
  dim3 block(THREADS);             // 256 threads = 8 waves (wave32)
  lstm_wmma_kernel<<<grid, block, 0, stream>>>(x, W_ih, W_hh, b_ih, b_hh, fc_w, fc_b, out);
}
